// GAT_Attention_37666863186195
// MI455X (gfx1250) — matrix operand order repaired and checked by reading, not compiled, not run on hardware
//
#include <hip/hip_runtime.h>
#include <hip/hip_bf16.h>

typedef __attribute__((ext_vector_type(16))) __bf16 v16bf;
typedef __attribute__((ext_vector_type(8)))  __bf16 v8bf;
typedef __attribute__((ext_vector_type(8)))  float  v8f;

#define NNODES 50000
#define KN 16
#define FDIM 128
#define NB 8            // nodes per block
#define ALPHA 0.2f
#define NEG_INF (-9.0e15f)

__device__ __forceinline__ unsigned short f32_bf16(float f) {
  unsigned int u = __float_as_uint(f);
  u += 0x7FFFu + ((u >> 16) & 1u);          // round-to-nearest-even
  return (unsigned short)(u >> 16);
}
__device__ __forceinline__ unsigned int pack_bf16x2(float a, float b) {
  return (unsigned int)f32_bf16(a) | ((unsigned int)f32_bf16(b) << 16);
}

// gfx1250 async copy: global -> LDS, 16B per lane, tracked by ASYNCcnt.
// GVS addressing: mem = SADDR(64b) + VADDR(32b offset); LDS dst = VDST vgpr.
__device__ __forceinline__ void async_b128_to_lds(unsigned int lds_addr,
                                                  unsigned int goff,
                                                  const void* gbase) {
  asm volatile("global_load_async_to_lds_b128 %0, %1, %2 offset:0"
               :: "v"(lds_addr), "v"(goff), "s"(gbase)
               : "memory");
}
__device__ __forceinline__ void wait_async0() {
  asm volatile("s_wait_asynccnt 0" ::: "memory");
}

// LDS layout (bytes):
//   lsW   : 32768  (128x128 bf16, B-operand swizzled)
//   lsA   : 32768  (128x128 bf16, x rows, row-major)
//   lsHj  : 65536  (128x128 f32)  -- ALSO used as f32 W staging before WMMA
//   lsH   :  8192  (16x128 f32, 8 valid rows)
//   lsO   :  4096  (16x128 bf16, orignal_x rows padded)
//   lsAv  :  1024  (a1|a2 f32)
//   lsS   :   512  (scores / attention, 8x16 f32)
//   lsHs  :    32  (per-node h-score)
//   lsAdj :   512  (8x16 int)
#define SMEM_BYTES (32768 + 32768 + 65536 + 8192 + 4096 + 1024 + 512 + 32 + 512)

__global__ __launch_bounds__(256)
void gat_fused_kernel(const float* __restrict__ gx,   // orignal_x (N,F)
                      const float* __restrict__ xk,   // x (N,K,F)
                      const int*   __restrict__ adj,  // (N,K)
                      const float* __restrict__ W,    // (F,F)
                      const float* __restrict__ av,   // (2F,1)
                      float* __restrict__ out)        // (N,F)
{
  extern __shared__ char smem[];
  unsigned short* lsW  = (unsigned short*)smem;                 // 16384 ushort
  unsigned short* lsA  = lsW + 16384;                           // 16384 ushort
  float*          lsHj = (float*)(lsA + 16384);                 // 16384 f32
  float*          lsH  = lsHj + 16384;                          // 2048 f32
  unsigned short* lsO  = (unsigned short*)(lsH + 2048);         // 2048 ushort
  float*          lsAv = (float*)(lsO + 2048);                  // 256 f32
  float*          lsS  = lsAv + 256;                            // 128 f32
  float*          lsHs = lsS + 128;                             // 8 f32
  int*            lsAdj= (int*)(lsHs + 8);                      // 128 int
  float*          stW  = lsHj;                                  // f32 W staging (dead until WMMA)

  const int tid   = threadIdx.x;
  const int lane  = tid & 31;
  const int wv    = tid >> 5;           // wave 0..7
  const int khalf = lane >> 4;
  const int l15   = lane & 15;
  const int nb0   = blockIdx.x * NB;

  // ---- kick off async W (f32, 64KB contiguous) -> LDS staging ----
  {
    unsigned int ldsbase = (unsigned int)(size_t)(void*)stW;
    #pragma unroll
    for (int i = 0; i < 16; ++i) {
      unsigned int off = (unsigned int)(tid * 16 + i * 4096);
      async_b128_to_lds(ldsbase + off, off, (const void*)W);
    }
  }

  // ---- stage a1|a2 and adj ----
  lsAv[tid] = av[tid];
  if (tid < NB * KN) {
    int node = nb0 + tid / KN;
    lsAdj[tid] = (node < NNODES) ? adj[node * KN + (tid % KN)] : 0;
  }

  // ---- x rows -> bf16 row-major in LDS (coalesced float4) ----
  {
    int r  = tid >> 1;                  // local row 0..127 = node*16 + k
    int fb = (tid & 1) * 64;
    int node = nb0 + (r >> 4);
    unsigned int* dst = (unsigned int*)(lsA + r * FDIM + fb);
    if (node < NNODES) {
      const float4* src = (const float4*)(xk + ((size_t)node * KN + (r & 15)) * FDIM + fb);
      #pragma unroll
      for (int i = 0; i < 16; ++i) {
        float4 v = src[i];
        dst[i * 2 + 0] = pack_bf16x2(v.x, v.y);
        dst[i * 2 + 1] = pack_bf16x2(v.z, v.w);
      }
    } else {
      #pragma unroll
      for (int i = 0; i < 32; ++i) dst[i] = 0u;
    }
  }

  // ---- orignal_x rows -> bf16 (pad to 16 rows) ----
  {
    int r  = tid >> 4;                  // 0..15
    int fb = (tid & 15) * 8;
    int node = nb0 + r;
    unsigned int* dst = (unsigned int*)(lsO + r * FDIM + fb);
    if (r < NB && node < NNODES) {
      const float4* src = (const float4*)(gx + (size_t)node * FDIM + fb);
      float4 v0 = src[0], v1 = src[1];
      dst[0] = pack_bf16x2(v0.x, v0.y);
      dst[1] = pack_bf16x2(v0.z, v0.w);
      dst[2] = pack_bf16x2(v1.x, v1.y);
      dst[3] = pack_bf16x2(v1.z, v1.w);
    } else {
      dst[0] = dst[1] = dst[2] = dst[3] = 0u;
    }
  }

  // prefetch next block's 64KB x tile into L2 (global_prefetch_b8)
  if (tid < 64) {
    int nnext = nb0 + NB;
    if (nnext < NNODES)
      __builtin_prefetch(xk + (size_t)nnext * KN * FDIM + tid * 256, 0, 1);
  }

  wait_async0();          // this wave's W chunks landed in LDS
  __syncthreads();        // everyone's chunks + x/O conversions visible

  // ---- W (f32 in LDS staging) -> bf16 pre-swizzled B-operand layout ----
  // b-frag for (kt,nt): lane = khalf*16 + ncol holds K = kt*32+khalf*16+j, N = nt*16+ncol
  #pragma unroll
  for (int c = 0; c < 4; ++c) {
    int combo = tid + 256 * c;          // 0..1023 = (kt, nt, lane)
    int kt  = combo >> 8;
    int rem = combo & 255;
    int nt  = rem >> 5;
    int ln  = rem & 31;
    int col = nt * 16 + (ln & 15);
    const float* wp = stW + (kt * 32 + (ln >> 4) * 16) * FDIM + col;
    unsigned int* dst = (unsigned int*)(lsW + (((kt * 8 + nt) * 32 + ln) << 4));
    #pragma unroll
    for (int j = 0; j < 16; j += 2)
      dst[j >> 1] = pack_bf16x2(wp[j * FDIM], wp[(j + 1) * FDIM]);
  }
  __syncthreads();        // lsW ready; staging area (lsHj) may now be overwritten

  // ---- WMMA: h_j tile (wave wv owns M-tile wv; 8 N-tiles x 4 K-steps) ----
  {
    const unsigned short* arow = lsA + (wv * 16 + l15) * FDIM;
    #pragma unroll
    for (int nt = 0; nt < 8; ++nt) {
      v8f acc = {};
      #pragma unroll
      for (int kt = 0; kt < 4; ++kt) {
        int ka = kt * 32 + khalf * 16;
        v8bf lo = *(const v8bf*)(arow + ka);
        v8bf hi = *(const v8bf*)(arow + ka + 8);
        v16bf afrag = __builtin_shufflevector(lo, hi, 0,1,2,3,4,5,6,7,8,9,10,11,12,13,14,15);
        v16bf bfrag = *(const v16bf*)(lsW + (((kt * 8 + nt) * 32 + lane) << 4));
        acc = __builtin_amdgcn_wmma_f32_16x16x32_bf16(false, afrag, false, bfrag,
                                                      (short)0, acc, false, false);
      }
      float* o = lsHj + (wv * 16 + khalf * 8) * FDIM + nt * 16 + l15;
      #pragma unroll
      for (int i = 0; i < 8; ++i) o[i * FDIM] = acc[i];
    }
  }

  // ---- WMMA: h tile (single M-tile of padded orignal rows; wave wv owns N-tile wv) ----
  {
    const unsigned short* arow = lsO + l15 * FDIM;
    v8f acc = {};
    #pragma unroll
    for (int kt = 0; kt < 4; ++kt) {
      int ka = kt * 32 + khalf * 16;
      v8bf lo = *(const v8bf*)(arow + ka);
      v8bf hi = *(const v8bf*)(arow + ka + 8);
      v16bf afrag = __builtin_shufflevector(lo, hi, 0,1,2,3,4,5,6,7,8,9,10,11,12,13,14,15);
      v16bf bfrag = *(const v16bf*)(lsW + (((kt * 8 + wv) * 32 + lane) << 4));
      acc = __builtin_amdgcn_wmma_f32_16x16x32_bf16(false, afrag, false, bfrag,
                                                    (short)0, acc, false, false);
    }
    float* o = lsH + (khalf * 8) * FDIM + wv * 16 + l15;
    #pragma unroll
    for (int i = 0; i < 8; ++i) o[i * FDIM] = acc[i];
  }
  __syncthreads();

  // ---- scores: hs[n] = h[n]·a1 ; s[n,k] = h_j[n,k]·a2 ----
  if (tid < NB) {
    const float4* hr = (const float4*)(lsH + tid * FDIM);
    const float4* a1 = (const float4*)(lsAv);
    float s = 0.f;
    #pragma unroll 8
    for (int i = 0; i < 32; ++i) {
      float4 h4 = hr[i], a4 = a1[i];
      s += h4.x * a4.x + h4.y * a4.y + h4.z * a4.z + h4.w * a4.w;
    }
    lsHs[tid] = s;
  }
  if (tid < NB * KN) {
    const float4* hj = (const float4*)(lsHj + tid * FDIM);
    const float4* a2 = (const float4*)(lsAv + FDIM);
    float s = 0.f;
    #pragma unroll 8
    for (int i = 0; i < 32; ++i) {
      float4 h4 = hj[i], a4 = a2[i];
      s += h4.x * a4.x + h4.y * a4.y + h4.z * a4.z + h4.w * a4.w;
    }
    lsS[tid] = s;
  }
  __syncthreads();

  // ---- leaky-relu + adjacency mask ----
  if (tid < NB * KN) {
    float e = lsHs[tid / KN] + lsS[tid];
    e = (e > 0.f) ? e : ALPHA * e;
    lsS[tid] = (lsAdj[tid] > 0) ? e : NEG_INF;
  }
  __syncthreads();

  // ---- softmax over K=16 per node ----
  if (tid < NB) {
    float* s = lsS + tid * KN;
    float m = s[0];
    #pragma unroll
    for (int k = 1; k < KN; ++k) m = fmaxf(m, s[k]);
    float w[KN];
    float sum = 0.f;
    #pragma unroll
    for (int k = 0; k < KN; ++k) { w[k] = __expf(s[k] - m); sum += w[k]; }
    float inv = 1.f / sum;
    #pragma unroll
    for (int k = 0; k < KN; ++k) s[k] = w[k] * inv;
  }
  __syncthreads();

  // ---- weighted sum + residual + ELU + store (float4 per thread) ----
  {
    int node = tid >> 5;                // 0..7
    int fb   = (tid & 31) * 4;
    int gn   = nb0 + node;
    if (gn < NNODES) {
      float4 hv = *(const float4*)(lsH + node * FDIM + fb);
      float r0 = hv.x, r1 = hv.y, r2 = hv.z, r3 = hv.w;
      #pragma unroll
      for (int k = 0; k < KN; ++k) {
        float w = lsS[node * KN + k];
        float4 hj = *(const float4*)(lsHj + (node * KN + k) * FDIM + fb);
        r0 += w * hj.x; r1 += w * hj.y; r2 += w * hj.z; r3 += w * hj.w;
      }
      float4 o;
      o.x = (r0 > 0.f) ? r0 : expm1f(r0);
      o.y = (r1 > 0.f) ? r1 : expm1f(r1);
      o.z = (r2 > 0.f) ? r2 : expm1f(r2);
      o.w = (r3 > 0.f) ? r3 : expm1f(r3);
      *(float4*)(out + (size_t)gn * FDIM + fb) = o;
    }
  }
}

extern "C" void kernel_launch(void* const* d_in, const int* in_sizes, int n_in,
                              void* d_out, int out_size, void* d_ws, size_t ws_size,
                              hipStream_t stream) {
  (void)in_sizes; (void)n_in; (void)out_size; (void)d_ws; (void)ws_size;
  const float* gx  = (const float*)d_in[0];   // orignal_x (N,F)
  const float* xk  = (const float*)d_in[1];   // x (N,K,F)
  const int*   adj = (const int*)d_in[2];     // adj (N,K)
  const float* W   = (const float*)d_in[3];   // W (F,F)
  const float* av  = (const float*)d_in[4];   // a (2F,1)
  float* out = (float*)d_out;

  const int blocks = (NNODES + NB - 1) / NB;  // 6250
  gat_fused_kernel<<<blocks, 256, SMEM_BYTES, stream>>>(gx, xk, adj, W, av, out);
}
